// VisionTransformer_41936060678209
// MI455X (gfx1250) — matrix operand order repaired and checked by reading, not compiled, not run on hardware
//
#include <hip/hip_runtime.h>
#include <hip/hip_bf16.h>
#include <math.h>

typedef __bf16 bf16;
typedef __attribute__((ext_vector_type(16))) __bf16 v16bf;
typedef __attribute__((ext_vector_type(8)))  __bf16 v8bf;
typedef __attribute__((ext_vector_type(8)))  float  v8f;

#define IMG    512
#define PSZ    16
#define DIM    768
#define NTOK   1025
#define NTP    1056            // padded token count: 66*16 == 33*32
#define BATCH  8
#define HEADS  12
#define HDIM   64
#define MLPD   3072
#define QKV3   2304
#define MROWS  (BATCH*NTP)     // 8448
#define PROWS  (BATCH*1024)    // 8192 patch rows
#define PDIMK  256             // 16*16*1 patch vector

// ---------------------------------------------------------------------------
// WMMA helpers (CDNA5: V_WMMA_F32_16X16X32_BF16, wave32)
// ---------------------------------------------------------------------------
__device__ __forceinline__ v8f wmma_bf16(v16bf a, v16bf b, v8f c) {
  // (neg_a, A, neg_b, B, c_mod, C, reuse_a, reuse_b)
  return __builtin_amdgcn_wmma_f32_16x16x32_bf16(false, a, false, b, (short)0, c, false, false);
}

// A-matrix 16x32 bf16 fragment: lane holds row M = lane%16.
// element e -> K = (lane/16)*16 + e (contiguous).
// base must already include  row*lda + k0 + (lane/16)*16
__device__ __forceinline__ v16bf load_a_frag(const bf16* base) {
  return *(const v16bf*)(base);
}

// B-matrix 32x16 bf16 fragment from a column-major (transposed) matrix:
// lane holds col N = lane%16; element e -> K = (lane/16)*16 + e (contiguous).
// base must already include  col*ldb(K) + k0 + (lane/16)*16  -> one 32B load.
__device__ __forceinline__ v16bf load_bT_frag(const bf16* base) {
  return *(const v16bf*)(base);
}

// ---------------------------------------------------------------------------
// Tiled GEMM:  C[M,N] = epilogue(A[M,K] @ B[K,N]),  B given TRANSPOSED [N][K].
// Block = 256 threads = 8 waves arranged 2(M) x 4(N); block tile 128x128.
// Wave tile 64x32: 4 M-subtiles x 2 N-subtiles = 8 WMMAs per K-step,
// all operands contiguous global loads (weights are L2-resident), no LDS.
// M % 128 == 0, N % 128 == 0, K % 32 == 0 (guaranteed by padding).
// ---------------------------------------------------------------------------
enum { MODE_PATCH = 0, MODE_BF16 = 1, MODE_BIAS_RES = 2, MODE_GELU_BF16 = 3 };

template <int MODE>
__global__ __launch_bounds__(256) void gemm_wmma(
    const bf16* __restrict__ A, const bf16* __restrict__ BT,
    float* __restrict__ outF, bf16* __restrict__ outB,
    const float* __restrict__ bias, const float* __restrict__ pos,
    int M, int N, int K)
{
  const int tid  = threadIdx.x;
  const int wave = tid >> 5;
  const int lane = tid & 31;
  const int half = lane >> 4;
  const int colN = lane & 15;
  const int wm   = wave >> 2;            // 0..1  (M group)
  const int wn   = wave & 3;             // 0..3  (N group)

  const int waveRow0 = blockIdx.x * 128 + wm * 64;
  const int waveCol0 = blockIdx.y * 128 + wn * 32;

  // per-subtile base pointers
  const bf16* aBase[4];
#pragma unroll
  for (int s = 0; s < 4; s++)
    aBase[s] = A + (size_t)(waveRow0 + s * 16 + colN) * K + half * 16;
  const bf16* bBase[2];
#pragma unroll
  for (int j = 0; j < 2; j++)
    bBase[j] = BT + (size_t)(waveCol0 + j * 16 + colN) * K + half * 16;

  v8f acc[4][2] = {};

  for (int k0 = 0; k0 < K; k0 += 32) {
    v16bf b0 = load_bT_frag(bBase[0] + k0);
    v16bf b1 = load_bT_frag(bBase[1] + k0);
#pragma unroll
    for (int s = 0; s < 4; s++) {
      v16bf af = load_a_frag(aBase[s] + k0);
      acc[s][0] = wmma_bf16(af, b0, acc[s][0]);
      acc[s][1] = wmma_bf16(af, b1, acc[s][1]);
    }
  }

  // ---------------- epilogues (C layout: row = r + 8*half, col = colN) -----
#pragma unroll
  for (int s = 0; s < 4; s++) {
#pragma unroll
    for (int j = 0; j < 2; j++) {
      const int col = waveCol0 + j * 16 + colN;
#pragma unroll
      for (int r = 0; r < 8; r++) {
        const int gr = waveRow0 + s * 16 + r + 8 * half;
        if (MODE == MODE_PATCH) {
          int bb = gr >> 10;
          int t  = 1 + (gr & 1023);              // token index (skip cls)
          size_t idx = ((size_t)(bb * NTP + t)) * DIM + col;
          outF[idx] = acc[s][j][r] + bias[col] + pos[(size_t)t * DIM + col];
        } else if (MODE == MODE_BF16) {
          outB[(size_t)gr * N + col] = (bf16)acc[s][j][r];
        } else if (MODE == MODE_BIAS_RES) {
          size_t idx = (size_t)gr * N + col;
          outF[idx] = outF[idx] + acc[s][j][r] + bias[col];
        } else {  // MODE_GELU_BF16 (exact gelu)
          float v = acc[s][j][r] + bias[col];
          outB[(size_t)gr * N + col] =
              (bf16)(0.5f * v * (1.0f + erff(v * 0.70710678118654752f)));
        }
      }
    }
  }
}

// ---------------------------------------------------------------------------
// Flash attention: one wave per (qtile, head, batch). Online softmax.
// qkv: bf16 [B, NTP, 2304] = [q | k | v] each head HDIM=64.
// ---------------------------------------------------------------------------
__global__ __launch_bounds__(32) void attn_flash(
    const bf16* __restrict__ qkv, bf16* __restrict__ obf)
{
  __shared__ __align__(64) bf16 Pl[16 * 32];   // P tile, C-layout -> A-layout relay

  const int lane = threadIdx.x;
  const int half = lane >> 4;
  const int colN = lane & 15;
  const int qt   = blockIdx.x;
  const int head = blockIdx.y;
  const int b    = blockIdx.z;
  const int q0   = qt * 16;

  // Q fragments: two 32-wide hdim chunks
  const bf16* qrow = qkv + ((size_t)(b * NTP + q0 + colN)) * QKV3 + head * HDIM + half * 16;
  v16bf aq0 = load_a_frag(qrow);
  v16bf aq1 = load_a_frag(qrow + 32);

  float m[8], l[8];
  v8f O0 = {}, O1 = {}, O2 = {}, O3 = {};
#pragma unroll
  for (int r = 0; r < 8; r++) { m[r] = -INFINITY; l[r] = 0.0f; }

  for (int kt = 0; kt < NTP / 32; kt++) {
    const int key0 = kt * 32;

    // ---- S = (Q @ K^T) for 32 keys: two 16-key subtiles ----
    v8f S0 = {}, S1 = {};
    {
      const bf16* kb = qkv + ((size_t)(b * NTP + key0 + colN)) * QKV3 + DIM + head * HDIM + half * 16;
      S0 = wmma_bf16(aq0, *(const v16bf*)(kb), S0);
      S0 = wmma_bf16(aq1, *(const v16bf*)(kb + 32), S0);
    }
    {
      const bf16* kb = qkv + ((size_t)(b * NTP + key0 + 16 + colN)) * QKV3 + DIM + head * HDIM + half * 16;
      S1 = wmma_bf16(aq0, *(const v16bf*)(kb), S1);
      S1 = wmma_bf16(aq1, *(const v16bf*)(kb + 32), S1);
    }

    float s0[8], s1[8];
    const bool mask0 = (key0 + colN) >= NTOK;
    const bool mask1 = (key0 + 16 + colN) >= NTOK;
#pragma unroll
    for (int r = 0; r < 8; r++) {
      s0[r] = mask0 ? -1.0e30f : S0[r] * 0.125f;   // scale = 1/sqrt(64)
      s1[r] = mask1 ? -1.0e30f : S1[r] * 0.125f;
    }

    // ---- online softmax (row = r + 8*half, shared by 16 lanes of a half) --
    float al[8];
#pragma unroll
    for (int r = 0; r < 8; r++) {
      float t = fmaxf(s0[r], s1[r]);
#pragma unroll
      for (int off = 1; off < 16; off <<= 1) t = fmaxf(t, __shfl_xor(t, off, 32));
      float mn = fmaxf(m[r], t);
      al[r] = __expf(m[r] - mn);
      m[r] = mn;
      s0[r] = __expf(s0[r] - mn);
      s1[r] = __expf(s1[r] - mn);
      float rs = s0[r] + s1[r];
#pragma unroll
      for (int off = 1; off < 16; off <<= 1) rs += __shfl_xor(rs, off, 32);
      l[r] = l[r] * al[r] + rs;
    }
#pragma unroll
    for (int r = 0; r < 8; r++) {
      O0[r] *= al[r]; O1[r] *= al[r]; O2[r] *= al[r]; O3[r] *= al[r];
    }

    // ---- P: C-layout -> A-layout via LDS ----
#pragma unroll
    for (int r = 0; r < 8; r++) {
      int rl = r + 8 * half;
      Pl[rl * 32 + colN]      = (bf16)s0[r];
      Pl[rl * 32 + 16 + colN] = (bf16)s1[r];
    }
    __syncthreads();
    v16bf pa = load_a_frag(&Pl[colN * 32 + half * 16]);

    // ---- V B-fragments (4 x 16 output cols) ----
    v16bf vb0, vb1, vb2, vb3;
#pragma unroll
    for (int e = 0; e < 16; e++) {
      const bf16* vr = qkv + ((size_t)(b * NTP + key0 + half * 16 + e)) * QKV3
                           + 2 * DIM + head * HDIM + colN;
      vb0[e] = vr[0]; vb1[e] = vr[16]; vb2[e] = vr[32]; vb3[e] = vr[48];
    }
    O0 = wmma_bf16(pa, vb0, O0);
    O1 = wmma_bf16(pa, vb1, O1);
    O2 = wmma_bf16(pa, vb2, O2);
    O3 = wmma_bf16(pa, vb3, O3);
    __syncthreads();
  }

  // ---- normalize & store (bf16, feeds out-proj GEMM) ----
#pragma unroll
  for (int r = 0; r < 8; r++) {
    float inv = 1.0f / l[r];
    int t = q0 + r + 8 * half;
    size_t base = ((size_t)(b * NTP + t)) * DIM + head * HDIM + colN;
    obf[base]      = (bf16)(O0[r] * inv);
    obf[base + 16] = (bf16)(O1[r] * inv);
    obf[base + 32] = (bf16)(O2[r] * inv);
    obf[base + 48] = (bf16)(O3[r] * inv);
  }
}

// ---------------------------------------------------------------------------
// LayerNorm -> bf16 (zeroes padded rows so padded tokens stay inert)
// ---------------------------------------------------------------------------
__global__ __launch_bounds__(256) void ln_bf16(
    const float* __restrict__ h, const float* __restrict__ g,
    const float* __restrict__ be, bf16* __restrict__ y)
{
  const int row = blockIdx.x;
  const int t = row % NTP;
  const float* hr = h + (size_t)row * DIM;
  bf16* yr = y + (size_t)row * DIM;
  if (t >= NTOK) {
    for (int c = threadIdx.x; c < DIM; c += 256) yr[c] = (bf16)0.0f;
    return;
  }
  __shared__ float r1[256], r2[256];
  float a = 0.0f, q = 0.0f;
  for (int c = threadIdx.x; c < DIM; c += 256) { float v = hr[c]; a += v; q += v * v; }
  r1[threadIdx.x] = a; r2[threadIdx.x] = q;
  __syncthreads();
  for (int off = 128; off > 0; off >>= 1) {
    if (threadIdx.x < off) { r1[threadIdx.x] += r1[threadIdx.x + off]; r2[threadIdx.x] += r2[threadIdx.x + off]; }
    __syncthreads();
  }
  float mean = r1[0] * (1.0f / DIM);
  float var  = r2[0] * (1.0f / DIM) - mean * mean;
  float rstd = rsqrtf(var + 1e-5f);
  for (int c = threadIdx.x; c < DIM; c += 256)
    yr[c] = (bf16)((hr[c] - mean) * rstd * g[c] + be[c]);
}

// ---------------------------------------------------------------------------
// misc small kernels
// ---------------------------------------------------------------------------
// f32 [K][N] row-major  ->  bf16 [N][K] (transposed, for contiguous B frags)
__global__ void f2bfT(const float* __restrict__ s, bf16* __restrict__ d,
                      int K, int N) {
  int i = blockIdx.x * 256 + threadIdx.x;
  if (i >= K * N) return;
  int k = i / N, n = i % N;
  d[(size_t)n * K + k] = (bf16)s[i];
}

__global__ void patchify(const float* __restrict__ x, bf16* __restrict__ p) {
  size_t i = (size_t)blockIdx.x * 256 + threadIdx.x;
  if (i >= (size_t)PROWS * PDIMK) return;
  int col = (int)(i & 255);
  size_t row = i >> 8;
  int b = (int)(row >> 10), pp = (int)(row & 1023);
  int hp = pp >> 5, wp = pp & 31;
  int p1 = col >> 4, p2 = col & 15;
  p[i] = (bf16)x[((size_t)b * IMG + hp * PSZ + p1) * IMG + wp * PSZ + p2];
}

__global__ void init_h(float* __restrict__ h, const float* __restrict__ cls,
                       const float* __restrict__ pos) {
  size_t i = (size_t)blockIdx.x * 256 + threadIdx.x;
  if (i >= (size_t)MROWS * DIM) return;
  int d = (int)(i % DIM);
  int t = (int)((i / DIM) % NTP);
  if (t == 0)          h[i] = cls[d] + pos[d];
  else if (t >= NTOK)  h[i] = 0.0f;
}

__global__ __launch_bounds__(256) void cls_head(
    const float* __restrict__ h, const float* __restrict__ g,
    const float* __restrict__ be, const float* __restrict__ w,
    const float* __restrict__ cb, float* __restrict__ out)
{
  const int b = blockIdx.x;
  const float* hr = h + (size_t)b * NTP * DIM;   // token 0 (cls)
  __shared__ float r1[256], r2[256];
  __shared__ float xn[DIM];
  float a = 0.0f, q = 0.0f;
  for (int c = threadIdx.x; c < DIM; c += 256) { float v = hr[c]; a += v; q += v * v; }
  r1[threadIdx.x] = a; r2[threadIdx.x] = q;
  __syncthreads();
  for (int off = 128; off > 0; off >>= 1) {
    if (threadIdx.x < off) { r1[threadIdx.x] += r1[threadIdx.x + off]; r2[threadIdx.x] += r2[threadIdx.x + off]; }
    __syncthreads();
  }
  float mean = r1[0] * (1.0f / DIM);
  float var  = r2[0] * (1.0f / DIM) - mean * mean;
  float rstd = rsqrtf(var + 1e-5f);
  for (int c = threadIdx.x; c < DIM; c += 256)
    xn[c] = (hr[c] - mean) * rstd * g[c] + be[c];
  __syncthreads();
  if (threadIdx.x < 10) {
    float acc = cb[threadIdx.x];
    for (int k = 0; k < DIM; k++) acc += xn[k] * w[k * 10 + threadIdx.x];
    out[b * 10 + threadIdx.x] = acc;
  }
}

// ---------------------------------------------------------------------------
// host orchestration
// ---------------------------------------------------------------------------
extern "C" void kernel_launch(void* const* d_in, const int* in_sizes, int n_in,
                              void* d_out, int out_size, void* d_ws, size_t ws_size,
                              hipStream_t stream)
{
  const float* x       = (const float*)d_in[0];
  const float* patch_w = (const float*)d_in[1];
  const float* patch_b = (const float*)d_in[2];
  const float* pos_emb = (const float*)d_in[3];
  const float* cls_tok = (const float*)d_in[4];
  const float* ln1_g   = (const float*)d_in[5];
  const float* ln1_b   = (const float*)d_in[6];
  const float* qkv_w   = (const float*)d_in[7];
  const float* out_w   = (const float*)d_in[8];
  const float* out_b   = (const float*)d_in[9];
  const float* ln2_g   = (const float*)d_in[10];
  const float* ln2_b   = (const float*)d_in[11];
  const float* mlp_w1  = (const float*)d_in[12];
  const float* mlp_b1  = (const float*)d_in[13];
  const float* mlp_w2  = (const float*)d_in[14];
  const float* mlp_b2  = (const float*)d_in[15];
  const float* lnf_g   = (const float*)d_in[16];
  const float* lnf_b   = (const float*)d_in[17];
  const float* clf_w   = (const float*)d_in[18];
  const float* clf_b   = (const float*)d_in[19];

  char* ws = (char*)d_ws;
  size_t off = 0;
  auto carve = [&](size_t bytes) -> char* {
    char* p = ws + off;
    off += (bytes + 255) & ~(size_t)255;
    return p;
  };
  float* h   = (float*)carve((size_t)MROWS * DIM * 4);
  bf16*  y   = (bf16*) carve((size_t)MROWS * DIM * 2);
  bf16*  qkv = (bf16*) carve((size_t)MROWS * QKV3 * 2);
  bf16*  obf = (bf16*) carve((size_t)MROWS * DIM * 2);
  bf16*  m1  = (bf16*) carve((size_t)MROWS * MLPD * 2);
  bf16*  pat = (bf16*) carve((size_t)PROWS * PDIMK * 2);
  bf16*  wbf = (bf16*) carve((size_t)MLPD * DIM * 2);   // largest weight panel

  // ---- patch embedding ----
  f2bfT<<<(PDIMK * DIM + 255) / 256, 256, 0, stream>>>(patch_w, wbf, PDIMK, DIM);
  patchify<<<(unsigned)(((size_t)PROWS * PDIMK + 255) / 256), 256, 0, stream>>>(x, pat);
  init_h<<<(unsigned)(((size_t)MROWS * DIM + 255) / 256), 256, 0, stream>>>(h, cls_tok, pos_emb);
  gemm_wmma<MODE_PATCH><<<dim3(PROWS / 128, DIM / 128), 256, 0, stream>>>(
      pat, wbf, h, nullptr, patch_b, pos_emb, PROWS, DIM, PDIMK);

  // ---- transformer blocks ----
  for (int l = 0; l < 12; l++) {
    ln_bf16<<<MROWS, 256, 0, stream>>>(h, ln1_g + l * DIM, ln1_b + l * DIM, y);

    f2bfT<<<(DIM * QKV3 + 255) / 256, 256, 0, stream>>>(qkv_w + (size_t)l * DIM * QKV3, wbf, DIM, QKV3);
    gemm_wmma<MODE_BF16><<<dim3(MROWS / 128, QKV3 / 128), 256, 0, stream>>>(
        y, wbf, nullptr, qkv, nullptr, nullptr, MROWS, QKV3, DIM);

    attn_flash<<<dim3(NTP / 16, HEADS, BATCH), 32, 0, stream>>>(qkv, obf);

    f2bfT<<<(DIM * DIM + 255) / 256, 256, 0, stream>>>(out_w + (size_t)l * DIM * DIM, wbf, DIM, DIM);
    gemm_wmma<MODE_BIAS_RES><<<dim3(MROWS / 128, DIM / 128), 256, 0, stream>>>(
        obf, wbf, h, nullptr, out_b + l * DIM, nullptr, MROWS, DIM, DIM);

    ln_bf16<<<MROWS, 256, 0, stream>>>(h, ln2_g + l * DIM, ln2_b + l * DIM, y);

    f2bfT<<<(DIM * MLPD + 255) / 256, 256, 0, stream>>>(mlp_w1 + (size_t)l * DIM * MLPD, wbf, DIM, MLPD);
    gemm_wmma<MODE_GELU_BF16><<<dim3(MROWS / 128, MLPD / 128), 256, 0, stream>>>(
        y, wbf, nullptr, m1, mlp_b1 + l * MLPD, nullptr, MROWS, MLPD, DIM);

    f2bfT<<<(MLPD * DIM + 255) / 256, 256, 0, stream>>>(mlp_w2 + (size_t)l * MLPD * DIM, wbf, MLPD, DIM);
    gemm_wmma<MODE_BIAS_RES><<<dim3(MROWS / 128, DIM / 128), 256, 0, stream>>>(
        m1, wbf, h, nullptr, mlp_b2 + l * DIM, nullptr, MROWS, DIM, MLPD);
  }

  // ---- final LN + classifier ----
  cls_head<<<BATCH, 256, 0, stream>>>(h, lnf_g, lnf_b, clf_w, clf_b, (float*)d_out);
}
